// MS_SSA_Conv_63565515980913
// MI455X (gfx1250) — matrix operand order repaired and checked by reading, not compiled, not run on hardware
//
#include <hip/hip_runtime.h>

typedef __attribute__((ext_vector_type(16))) _Float16 v16h;
typedef __attribute__((ext_vector_type(8)))  _Float16 v8h;
typedef __attribute__((ext_vector_type(8)))  float    v8f;

#define Tt 4
#define Bb 16
#define Cc 256
#define Nn 1024
#define HEADS 8
#define HD 32
#define BCN (Bb*Cc*Nn)          /* 4,194,304 */
#define TBCN ((size_t)Tt*BCN)   /* 16,777,216 */

__device__ __forceinline__ v16h cat8(v8h lo, v8h hi) {
  return __builtin_shufflevector(lo, hi, 0,1,2,3,4,5,6,7,8,9,10,11,12,13,14,15);
}

// ---------------- weight prep: fold BN scale/shift into f16 weights ----------------
__global__ __launch_bounds__(256)
void prep_kern(const float* __restrict__ w, const float* __restrict__ g,
               const float* __restrict__ bt, const float* __restrict__ mn,
               const float* __restrict__ vr, const float* __restrict__ extrabias,
               _Float16* __restrict__ weff, float* __restrict__ beff) {
  int o = blockIdx.x;      // out channel
  int c = threadIdx.x;     // in channel
  float scale = g[o] / sqrtf(vr[o] + 1e-5f);
  float shift = bt[o] - mn[o] * scale;
  weff[o*Cc + c] = (_Float16)(scale * w[o*Cc + c]);
  if (c == 0) beff[o] = shift + (extrabias ? scale * extrabias[o] : 0.0f);
}

// ---------------- LIF over x: (T,B,C,N) -> binary f16 spikes ----------------
__global__ __launch_bounds__(256)
void lif_x_kern(const float* __restrict__ x, _Float16* __restrict__ xs) {
  size_t gid = (size_t)blockIdx.x * 256 + threadIdx.x;   // over (b,c,n)
  float v = 0.0f;
  #pragma unroll
  for (int t = 0; t < Tt; ++t) {
    float xt = x[(size_t)t*BCN + gid];
    float h = 0.5f * (v + xt);
    float s = (h >= 1.0f) ? 1.0f : 0.0f;
    v = h * (1.0f - s);
    xs[(size_t)t*BCN + gid] = (_Float16)s;
  }
}

// ---------------- LIF over q/k/v pre-activations (in-place -> spikes) ----------------
__global__ __launch_bounds__(256)
void lif_qkv_kern(_Float16* __restrict__ q, _Float16* __restrict__ k,
                  _Float16* __restrict__ vb, float* __restrict__ vout) {
  size_t gid = (size_t)blockIdx.x * 256 + threadIdx.x;   // over (b,c,n)
  int n = (int)(gid & 1023);
  int c = (int)((gid >> 10) & 255);
  int b = (int)(gid >> 18);
  int hh = c >> 5, d = c & 31;
  float vq = 0.0f, vk = 0.0f, vv = 0.0f;
  #pragma unroll
  for (int t = 0; t < Tt; ++t) {
    size_t idx = (size_t)t*BCN + gid;
    float h, s;
    h = 0.5f * (vq + (float)q[idx]); s = (h >= 1.0f) ? 1.0f : 0.0f;
    vq = h * (1.0f - s); q[idx] = (_Float16)s;
    h = 0.5f * (vk + (float)k[idx]); s = (h >= 1.0f) ? 1.0f : 0.0f;
    vk = h * (1.0f - s); k[idx] = (_Float16)s;
    h = 0.5f * (vv + (float)vb[idx]); s = (h >= 1.0f) ? 1.0f : 0.0f;
    vv = h * (1.0f - s); vb[idx] = (_Float16)s;
    // v spikes also go to d_out in to_heads layout (T,B,heads,N,hd)
    vout[(((size_t)t*Bb + b)*HEADS + hh)*((size_t)Nn*HD) + (size_t)n*HD + d] = s;
  }
}

// ---------------- kv = sum over N of k*v, per (t,b,head,hd) ----------------
__global__ __launch_bounds__(256)
void kv_reduce_kern(const _Float16* __restrict__ ks, const _Float16* __restrict__ vs,
                    float* __restrict__ kv_sum) {
  int bx = blockIdx.x;                 // (t,b,hh)
  int t = bx >> 7, b = (bx >> 3) & 15, hh = bx & 7;
  int tid = threadIdx.x;
  int d = tid & 31, i = tid >> 5;      // 8 partials per d
  int c = hh*HD + d;
  size_t base = (((size_t)t*Bb + b)*Cc + c)*Nn;
  float s = 0.0f;
  for (int n = i*128; n < i*128 + 128; ++n)
    s += (float)ks[base + n] * (float)vs[base + n];
  __shared__ float red[256];
  red[tid] = s;
  __syncthreads();
  if (tid < 32) {
    float tot = 0.0f;
    #pragma unroll
    for (int j = 0; j < 8; ++j) tot += red[j*32 + tid];
    kv_sum[(((size_t)t*Bb + b)*HEADS + hh)*HD + tid] = tot;
  }
}

// ---------------- LIF on kv (v_th = 0.5) ----------------
__global__ __launch_bounds__(256)
void kv_lif_kern(const float* __restrict__ kv_sum, float* __restrict__ kv_spike) {
  int gid = blockIdx.x * 256 + threadIdx.x;   // over (b,hh,d): 4096
  float v = 0.0f;
  #pragma unroll
  for (int t = 0; t < Tt; ++t) {
    float xt = kv_sum[t*4096 + gid];
    float h = 0.5f * (v + xt);
    float s = (h >= 0.5f) ? 1.0f : 0.0f;
    v = h * (1.0f - s);
    kv_spike[t*4096 + gid] = s;
  }
}

// ---------------- attn = q * kv (broadcast over N), write f16 ----------------
__global__ __launch_bounds__(256)
void attn_kern(const _Float16* __restrict__ qs, const float* __restrict__ kv_spike,
               _Float16* __restrict__ attn) {
  size_t gid = (size_t)blockIdx.x * 256 + threadIdx.x;   // over (t,b,c,n)
  int c  = (int)((gid >> 10) & 255);
  int tb = (int)(gid >> 18);
  float kvv = kv_spike[(size_t)tb*Cc + c];   // hh*32+d == c
  attn[gid] = (_Float16)((float)qs[gid] * kvv);
}

// ---------------- WMMA GEMM: Y(256x1024) = W'(256x256) x X(256x1024) per (t,b) ----
// workgroup = 8 waves; output tile 128 rows x 64 cols; K staged in 2 chunks of 128.
// A panel staged with CDNA5 async VMEM->LDS DMA (ASYNCcnt); B panel transposed via VGPRs.
__global__ __launch_bounds__(256)
void gemm256(const _Float16* __restrict__ W,    // 256x256 row-major f16 (BN folded)
             const float* __restrict__ bias,    // per-row shift (256)
             const _Float16* __restrict__ Xin,  // (T*B, 256, 1024) f16
             _Float16* __restrict__ outH,       // f16 out (qkv pre), or null
             float* __restrict__ outF,          // f32 out (final), or null
             const float* __restrict__ ident) { // identity add, or null
  const int tb = blockIdx.x;            // 0..63  (t*B+b)
  const int o0 = blockIdx.y * 128;      // 0 or 128
  const int n0 = blockIdx.z * 64;       // 0..960
  const int tid  = threadIdx.x;
  const int wave = tid >> 5;
  const int lane = tid & 31;

  __shared__ _Float16 As[128*128];      // [row][k]  (32 KB)
  __shared__ _Float16 Bs[64*128];       // [n][k]    (16 KB)

  const _Float16* Xtb = Xin + (size_t)tb * Cc * Nn;

  v8f acc[4] = {v8f{}, v8f{}, v8f{}, v8f{}};

  for (int kc = 0; kc < 2; ++kc) {
    // stage A via async global->LDS B128 copies: rows o0..o0+127, K chunk kc*128..+127
    // (contiguous both sides -> pure DMA, no VGPR round trip; tracked by ASYNCcnt)
    #pragma unroll
    for (int i = 0; i < 8; ++i) {
      int v = tid + 256*i;              // 0..2047
      int row = v >> 4;
      int k8  = (v & 15) << 3;
      unsigned ldsoff = (unsigned)(size_t)(const void*)(As + row*128 + k8);
      const _Float16* gp = W + (size_t)(o0 + row)*Cc + kc*128 + k8;
      asm volatile("global_load_async_to_lds_b128 %0, %1, off"
                   :: "v"(ldsoff), "v"(gp) : "memory");
    }
    // stage B transposed: Bs[n][k] = X[kc*128+k][n0+n]
    #pragma unroll
    for (int i = 0; i < 4; ++i) {
      int v = tid + 256*i;              // 0..1023
      int k   = v & 127;
      int nch = v >> 7;                 // 0..7
      v8h dv = *(const v8h*)(Xtb + (size_t)(kc*128 + k)*Nn + n0 + nch*8);
      #pragma unroll
      for (int j = 0; j < 8; ++j)
        Bs[(nch*8 + j)*128 + k] = dv[j];
    }
    if (kc == 0)  // hint next K chunk toward caches (global_prefetch_b8)
      __builtin_prefetch(Xtb + (size_t)(128 + (tid & 127))*Nn + n0, 0, 1);
    // async LDS writes must be complete (per-wave) before the workgroup barrier
    asm volatile("s_wait_asynccnt 0x0" ::: "memory");
    __syncthreads();

    #pragma unroll
    for (int ki = 0; ki < 4; ++ki) {
      // A fragment: lane m=l%16; lanes0-15 K={0..7,16..23}, lanes16-31 K={8..15,24..31}
      int m = lane & 15;
      int abase = (lane < 16) ? 0 : 8;
      const _Float16* ap = As + (wave*16 + m)*128 + ki*32 + abase;
      v16h a = cat8(*(const v8h*)(ap), *(const v8h*)(ap + 16));
      int kb = (lane < 16) ? 0 : 8;
      #pragma unroll
      for (int j = 0; j < 4; ++j) {
        // B fragment: lane col n=l%16; lanes0-15 K={0..7,16..23}, lanes16-31 K={8..15,24..31}
        const _Float16* bp = Bs + (j*16 + (lane & 15))*128 + ki*32 + kb;
        v16h bfrag = cat8(*(const v8h*)(bp), *(const v8h*)(bp + 16));
        acc[j] = __builtin_amdgcn_wmma_f32_16x16x32_f16(
            false, a, false, bfrag, (short)0, acc[j], false, false);
      }
    }
    __syncthreads();
  }

  // epilogue: C/D layout — VGPR r: lanes0-15 (M=r,N=lane), lanes16-31 (M=8+r,N=lane-16)
  #pragma unroll
  for (int j = 0; j < 4; ++j) {
    #pragma unroll
    for (int r = 0; r < 8; ++r) {
      int m = (lane < 16) ? r : (r + 8);
      int o = o0 + wave*16 + m;
      int n = n0 + j*16 + (lane & 15);
      float val = acc[j][r] + (bias ? bias[o] : 0.0f);
      size_t oidx = ((size_t)tb*Cc + o)*Nn + n;
      if (ident) val += ident[oidx];
      if (outF) outF[oidx] = val;
      else      outH[oidx] = (_Float16)val;
    }
  }
}

extern "C" void kernel_launch(void* const* d_in, const int* in_sizes, int n_in,
                              void* d_out, int out_size, void* d_ws, size_t ws_size,
                              hipStream_t stream) {
  (void)in_sizes; (void)n_in; (void)out_size; (void)ws_size;
  const float* x      = (const float*)d_in[0];
  const float* q_w    = (const float*)d_in[1];
  const float* q_g    = (const float*)d_in[2];
  const float* q_b    = (const float*)d_in[3];
  const float* q_m    = (const float*)d_in[4];
  const float* q_var  = (const float*)d_in[5];
  const float* k_w    = (const float*)d_in[6];
  const float* k_g    = (const float*)d_in[7];
  const float* k_b    = (const float*)d_in[8];
  const float* k_m    = (const float*)d_in[9];
  const float* k_var  = (const float*)d_in[10];
  const float* v_w    = (const float*)d_in[11];
  const float* v_g    = (const float*)d_in[12];
  const float* v_b    = (const float*)d_in[13];
  const float* v_m    = (const float*)d_in[14];
  const float* v_var  = (const float*)d_in[15];
  const float* proj_w = (const float*)d_in[16];
  const float* proj_bias = (const float*)d_in[17];
  const float* p_g    = (const float*)d_in[18];
  const float* p_b    = (const float*)d_in[19];
  const float* p_m    = (const float*)d_in[20];
  const float* p_var  = (const float*)d_in[21];

  // ---- workspace layout (~128.6 MiB total; all intermediates L2-resident) ----
  const size_t SPB = TBCN * sizeof(_Float16);   // 33,554,432 B per f16 spike/pre buffer
  char* ws = (char*)d_ws;
  _Float16* xs   = (_Float16*)(ws + 0);         // xs spikes; later reused for attn
  _Float16* qpre = (_Float16*)(ws + 1*SPB);     // q pre -> q spikes (in place)
  _Float16* kpre = (_Float16*)(ws + 2*SPB);     // k pre -> k spikes
  _Float16* vpre = (_Float16*)(ws + 3*SPB);     // v pre -> v spikes
  char* aux = ws + 4*SPB;
  const size_t WB = (size_t)Cc*Cc*sizeof(_Float16);  // 131072 B
  _Float16* wq = (_Float16*)(aux + 0*WB);
  _Float16* wk = (_Float16*)(aux + 1*WB);
  _Float16* wv = (_Float16*)(aux + 2*WB);
  _Float16* wp = (_Float16*)(aux + 3*WB);
  float* bq = (float*)(aux + 4*WB);
  float* bk = bq + Cc;
  float* bv = bq + 2*Cc;
  float* bp = bq + 3*Cc;
  float* kv_sum   = bq + 4*Cc;                  // T*B*HEADS*HD = 16384 floats
  float* kv_spike = kv_sum + 16384;
  _Float16* attn = xs;                          // reuse xs region after qkv GEMMs

  float* out0 = (float*)d_out;                  // (T,B,C,H,W) f32
  float* vout = out0 + TBCN;                    // (T,B,heads,N,hd) f32

  // 1) fold BN into f16 weights / f32 biases
  prep_kern<<<256, 256, 0, stream>>>(q_w, q_g, q_b, q_m, q_var, nullptr, wq, bq);
  prep_kern<<<256, 256, 0, stream>>>(k_w, k_g, k_b, k_m, k_var, nullptr, wk, bk);
  prep_kern<<<256, 256, 0, stream>>>(v_w, v_g, v_b, v_m, v_var, nullptr, wv, bv);
  prep_kern<<<256, 256, 0, stream>>>(proj_w, p_g, p_b, p_m, p_var, proj_bias, wp, bp);

  // 2) shortcut LIF -> xs spikes
  lif_x_kern<<<BCN/256, 256, 0, stream>>>(x, xs);

  // 3) q/k/v GEMMs (WMMA f16, BN folded)
  dim3 gg(Tt*Bb, Cc/128, Nn/64);   // (64, 2, 16)
  gemm256<<<gg, 256, 0, stream>>>(wq, bq, xs, qpre, nullptr, nullptr);
  gemm256<<<gg, 256, 0, stream>>>(wk, bk, xs, kpre, nullptr, nullptr);
  gemm256<<<gg, 256, 0, stream>>>(wv, bv, xs, vpre, nullptr, nullptr);

  // 4) LIF on q/k/v (in place); v spikes also to d_out (to_heads layout)
  lif_qkv_kern<<<BCN/256, 256, 0, stream>>>(qpre, kpre, vpre, vout);

  // 5) kv = sum_N k*v ; LIF(v_th=0.5)
  kv_reduce_kern<<<Tt*Bb*HEADS, 256, 0, stream>>>(kpre, vpre, kv_sum);
  kv_lif_kern<<<16, 256, 0, stream>>>(kv_sum, kv_spike);

  // 6) attn = q * kv (broadcast over N)
  attn_kern<<<(unsigned)(TBCN/256), 256, 0, stream>>>(qpre, kv_spike, attn);

  // 7) proj GEMM + bias + BN (folded) + identity -> d_out
  gemm256<<<gg, 256, 0, stream>>>(wp, bp, attn, nullptr, out0, x);
}
